// GAT_LSTM_34583076667750
// MI455X (gfx1250) — matrix operand order repaired and checked by reading, not compiled, not run on hardware
//
#include <hip/hip_runtime.h>

// ---------------------------------------------------------------------------
// Types / WMMA helpers (CDNA5 gfx1250, wave32)
// ---------------------------------------------------------------------------
typedef _Float16 h16;
typedef __attribute__((ext_vector_type(16))) _Float16 v16h;
typedef __attribute__((ext_vector_type(8)))  _Float16 v8h;
typedef __attribute__((ext_vector_type(8)))  float    v8f;
typedef __attribute__((ext_vector_type(4)))  float    v4f;

#define WMMA(a, b, c) \
  __builtin_amdgcn_wmma_f32_16x16x32_f16(false, (a), false, (b), (short)0, (c), false, false)

// A fragment (16 rows x 32 K) from row-major LDS tile, vectorized:
// lane 0-15 -> row=lane, needs halves [kb..kb+8) and [kb+8..kb+16), kb = 16*(lane>>4).
// Both groups are 16B-aligned contiguous runs -> 2x ds_load_b128.
__device__ __forceinline__ v16h frag_a(const h16* p, int ld) {
  int lane = threadIdx.x & 31;
  int row  = lane & 15;
  int kb   = (lane >> 4) * 16;
  const h16* q = p + row * ld + kb;
  v8h lo = *(const v8h*)(q);
  v8h hi = *(const v8h*)(q + 8);
  return __builtin_shufflevector(lo, hi, 0, 1, 2, 3, 4, 5, 6, 7, 8, 9, 10, 11, 12, 13, 14, 15);
}

// B fragment from swizzled (fragment-major) LDS: 16 contiguous halves per lane
// at [(ktile*ntiles + ntile)*32 + lane]*16 -> one v16h load (2x ds_load_b128).
__device__ __forceinline__ v16h frag_b_sw(const h16* Bsw, int ktile, int ntile, int ntiles) {
  int lane = threadIdx.x & 31;
  return *(const v16h*)(Bsw + (size_t)(((ktile * ntiles) + ntile) * 32 + lane) * 16);
}

__device__ __forceinline__ float sigmoidf(float x) { return 1.f / (1.f + __expf(-x)); }

// order-preserving float->uint key for atomicMax-based segment max
__device__ __forceinline__ unsigned fkey(float f) {
  unsigned u = __float_as_uint(f);
  return (u & 0x80000000u) ? ~u : (u | 0x80000000u);
}
__device__ __forceinline__ float funkey(unsigned u) {
  u = (u & 0x80000000u) ? (u & 0x7FFFFFFFu) : ~u;
  return __uint_as_float(u);
}

// ---------------------------------------------------------------------------
// Generic WMMA GEMM:  C[M,N] = A[M,K](f16) * B[K,N](f16) + bias, opt ReLU.
// Block = 128 threads (4 waves), 128x64 output tile; each wave owns TWO
// 16-row M-subtiles so every B fragment feeds two WMMAs. K chunked by 128.
// ---------------------------------------------------------------------------
#define TM 128
#define TN 64
#define KC 128
#define ALD (KC + 8)  // A tile leading dim (halves); 136*2B = 272B, 16B-aligned rows

__global__ void k_gemm_f16(const h16* __restrict__ A, const h16* __restrict__ B,
                           const float* __restrict__ bias, float* __restrict__ Cf,
                           h16* __restrict__ Ch, int M, int N, int K, int relu) {
  __shared__ __align__(32) h16 As[TM][ALD];
  __shared__ __align__(32) h16 Bsw[4 * 4 * 32 * 16];  // [ktile][ntile][lane][16]
  const int tid = threadIdx.x, wave = tid >> 5, lane = tid & 31;
  const int m0 = blockIdx.x * TM, n0 = blockIdx.y * TN;

  const v8h z8 = {0, 0, 0, 0, 0, 0, 0, 0};
  v8f zero = {0.f, 0.f, 0.f, 0.f, 0.f, 0.f, 0.f, 0.f};
  v8f acc[2][4];
#pragma unroll
  for (int mt = 0; mt < 2; ++mt)
#pragma unroll
    for (int i = 0; i < 4; ++i) acc[mt][i] = zero;

  for (int k0 = 0; k0 < K; k0 += KC) {
    int kc  = K - k0; if (kc > KC) kc = KC;
    int kcp = (kc + 31) & ~31;  // pad to WMMA K granularity, zero-filled

    // stage A: 8-half (16B) vector chunks; all Ks are multiples of 8
    int chunks = kcp >> 3;
    for (int i = tid; i < TM * chunks; i += 128) {
      int r = i / chunks, c8 = (i - r * chunks) * 8;
      int gr = m0 + r, gc = k0 + c8;
      v8h v = z8;
      if (gr < M && gc < K) v = *(const v8h*)(A + (size_t)gr * K + gc);
      *(v8h*)&As[r][c8] = v;
    }
    // stage B swizzled: element (k,n) -> [ktile][ntile][lane=(n&15)+16*(kin>>4)][j=kin&15]
    for (int i = tid; i < kcp * TN; i += 128) {
      int k = i >> 6, n = i & 63;
      int gk = k0 + k, gn = n0 + n;
      h16 v = (gk < K && gn < N) ? B[(size_t)gk * N + gn] : (h16)0;
      int ktile = k >> 5, kin = k & 31;
      int lidx  = (n & 15) + ((kin >> 4) << 4);
      Bsw[(size_t)((((ktile << 2) | (n >> 4)) << 5) | lidx) * 16 + (kin & 15)] = v;
    }
    __syncthreads();

    for (int kt = 0; kt < (kcp >> 5); ++kt) {
      v16h a0 = frag_a(&As[wave * 16][kt * 32], ALD);
      v16h a1 = frag_a(&As[64 + wave * 16][kt * 32], ALD);
#pragma unroll
      for (int nt = 0; nt < 4; ++nt) {
        v16h b = frag_b_sw(Bsw, kt, nt, 4);
        acc[0][nt] = WMMA(a0, b, acc[0][nt]);
        acc[1][nt] = WMMA(a1, b, acc[1][nt]);
      }
    }
    __syncthreads();
  }

  const int colLane = lane & 15;
  const int rbase   = (lane >> 4) * 8;  // C layout: VGPR r -> M = r (+8 for lanes 16-31)
#pragma unroll
  for (int mt = 0; mt < 2; ++mt) {
#pragma unroll
    for (int nt = 0; nt < 4; ++nt) {
      int gn = n0 + nt * 16 + colLane;
      float bv = (bias != nullptr && gn < N) ? bias[gn] : 0.f;
#pragma unroll
      for (int r = 0; r < 8; ++r) {
        int gm = m0 + mt * 64 + wave * 16 + rbase + r;
        if (gm < M && gn < N) {
          float v = acc[mt][nt][r] + bv;
          if (relu) v = fmaxf(v, 0.f);
          size_t idx = (size_t)gm * N + gn;
          if (Cf) Cf[idx] = v;
          if (Ch) Ch[idx] = (h16)v;
        }
      }
    }
  }
}

// ---------------------------------------------------------------------------
// Fused LSTM step: gates = X@WihT + H@WhhT + b -> pointwise -> C, Hout(f16).
// X rows strided by xStride (halves). WihT/WhhT are [128][512] f16 (pre-transposed).
// 128 rows per block; each wave owns two 16-row M-subtiles. Per hid-tile jn,
// the 4 gate accumulators (i,f,g,o) share the C layout, so each lane owns
// matching (row,hid) gate values for the pointwise stage.
// ---------------------------------------------------------------------------
__global__ void k_lstm_step(const h16* __restrict__ X, int xStride,
                            const h16* __restrict__ H, float* __restrict__ C,
                            h16* __restrict__ Hout,
                            const h16* __restrict__ WihT, const h16* __restrict__ WhhT,
                            const float* __restrict__ bih, const float* __restrict__ bhh,
                            int M) {
  __shared__ __align__(32) h16 Xs[128][136];
  __shared__ __align__(32) h16 Hs[128][136];
  __shared__ __align__(32) h16 Wi[4 * 4 * 32 * 16];  // swizzled fragment-major
  __shared__ __align__(32) h16 Wh[4 * 4 * 32 * 16];
  const int tid = threadIdx.x, wave = tid >> 5, lane = tid & 31;
  const int m0 = blockIdx.x * 128;
  const int colLane = lane & 15, rbase = (lane >> 4) * 8;
  const v8h z8 = {0, 0, 0, 0, 0, 0, 0, 0};

  // stage X,H: 128 rows x 16 chunks of 8 halves (16B vector ops)
  for (int i = tid; i < 128 * 16; i += 128) {
    int r = i >> 4, c8 = (i & 15) * 8;
    int gr = m0 + r;
    v8h vx = z8, vh = z8;
    if (gr < M) {
      vx = *(const v8h*)(X + (size_t)gr * xStride + c8);
      vh = *(const v8h*)(H + (size_t)gr * 128 + c8);
    }
    *(v8h*)&Xs[r][c8] = vx;
    *(v8h*)&Hs[r][c8] = vh;
  }

  v8f zero = {0.f, 0.f, 0.f, 0.f, 0.f, 0.f, 0.f, 0.f};

  for (int jn = 0; jn < 8; ++jn) {
    __syncthreads();  // previous iteration's fragment reads done before restage
    // stage Wih/Whh columns for the 4 gates of hid-tile jn (swizzled):
    // gate g -> global cols [g*128 + jn*16, +16)
    for (int i = tid; i < 128 * 64; i += 128) {
      int k = i >> 6, c = i & 63;
      int g = c >> 4, cc = c & 15;
      int gn = g * 128 + jn * 16 + cc;
      h16 vi = WihT[(size_t)k * 512 + gn];
      h16 vh = WhhT[(size_t)k * 512 + gn];
      int ktile = k >> 5, kin = k & 31;
      size_t idx = (size_t)((((ktile << 2) | g) << 5) | (cc + ((kin >> 4) << 4))) * 16 + (kin & 15);
      Wi[idx] = vi;
      Wh[idx] = vh;
    }
    __syncthreads();

    v8f acc[2][4];
#pragma unroll
    for (int mt = 0; mt < 2; ++mt)
#pragma unroll
      for (int g = 0; g < 4; ++g) acc[mt][g] = zero;

    for (int kt = 0; kt < 4; ++kt) {
      v16h ax0 = frag_a(&Xs[wave * 16][kt * 32], 136);
      v16h ah0 = frag_a(&Hs[wave * 16][kt * 32], 136);
      v16h ax1 = frag_a(&Xs[64 + wave * 16][kt * 32], 136);
      v16h ah1 = frag_a(&Hs[64 + wave * 16][kt * 32], 136);
#pragma unroll
      for (int g = 0; g < 4; ++g) {
        v16h bi = frag_b_sw(Wi, kt, g, 4);
        v16h bh = frag_b_sw(Wh, kt, g, 4);
        acc[0][g] = WMMA(ax0, bi, acc[0][g]);
        acc[1][g] = WMMA(ax1, bi, acc[1][g]);
        acc[0][g] = WMMA(ah0, bh, acc[0][g]);
        acc[1][g] = WMMA(ah1, bh, acc[1][g]);
      }
    }

    int hid = jn * 16 + colLane;
    float b_i = bih[0 * 128 + hid] + bhh[0 * 128 + hid];
    float b_f = bih[1 * 128 + hid] + bhh[1 * 128 + hid];
    float b_g = bih[2 * 128 + hid] + bhh[2 * 128 + hid];
    float b_o = bih[3 * 128 + hid] + bhh[3 * 128 + hid];
#pragma unroll
    for (int mt = 0; mt < 2; ++mt) {
#pragma unroll
      for (int r = 0; r < 8; ++r) {
        int gm = m0 + mt * 64 + wave * 16 + rbase + r;
        if (gm < M) {
          float ig = sigmoidf(acc[mt][0][r] + b_i);
          float fg = sigmoidf(acc[mt][1][r] + b_f);
          float gg = tanhf(acc[mt][2][r] + b_g);
          float og = sigmoidf(acc[mt][3][r] + b_o);
          size_t idx = (size_t)gm * 128 + hid;
          float c = fg * C[idx] + ig * gg;
          C[idx]    = c;
          Hout[idx] = (h16)(og * tanhf(c));
        }
      }
    }
  }
}

// ---------------------------------------------------------------------------
// GAT edge kernels
// ---------------------------------------------------------------------------
__global__ void k_gat_init(float* __restrict__ agg, const float* __restrict__ bias,
                           unsigned* __restrict__ maxk, float* __restrict__ den, int nAgg,
                           int nSeg) {
  int i = blockIdx.x * blockDim.x + threadIdx.x;
  if (i < nAgg) agg[i] = bias[i & 511];
  if (i < nSeg) { maxk[i] = 0u; den[i] = 0.f; }
}

// per-edge wave: logits[e,h] = dot(leaky_relu(xr[dst]+xl[src]), att[h]); seg-max via atomicMax
// 32 lanes x float4 = exactly 128 dims per head.
__global__ void k_edge_logit_max(const float* __restrict__ xl, const float* __restrict__ xr,
                                 const int* __restrict__ src, const int* __restrict__ dst,
                                 const float* __restrict__ att, float* __restrict__ logits,
                                 unsigned* __restrict__ maxk, int E) {
  int wave = threadIdx.x >> 5, lane = threadIdx.x & 31;
  int e = blockIdx.x * 4 + wave;
  if (e >= E) return;
  int s = src[e], d = dst[e];
  for (int h = 0; h < 4; ++h) {
    v4f vj = ((const v4f*)(xl + (size_t)s * 512 + h * 128))[lane];
    v4f vi = ((const v4f*)(xr + (size_t)d * 512 + h * 128))[lane];
    v4f va = ((const v4f*)(att + h * 128))[lane];
    float acc = 0.f;
#pragma unroll
    for (int q = 0; q < 4; ++q) {
      float v = vi[q] + vj[q];
      v = (v > 0.f) ? v : v * 0.2f;
      acc += v * va[q];
    }
#pragma unroll
    for (int off = 16; off > 0; off >>= 1) acc += __shfl_xor(acc, off, 32);
    if (lane == 0) {
      logits[(size_t)e * 4 + h] = acc;
      atomicMax(maxk + (size_t)d * 4 + h, fkey(acc));
    }
  }
}

__global__ void k_edge_exp(const float* __restrict__ logits, const unsigned* __restrict__ maxk,
                           const int* __restrict__ dst, float* __restrict__ ex,
                           float* __restrict__ den, int E) {
  int i = blockIdx.x * blockDim.x + threadIdx.x;
  if (i >= E * 4) return;
  int e = i >> 2, h = i & 3;
  int d = dst[e];
  float m = funkey(maxk[(size_t)d * 4 + h]);
  float v = __expf(logits[i] - m);
  ex[i] = v;
  atomicAdd(den + (size_t)d * 4 + h, v);
}

__global__ void k_edge_scatter(const float* __restrict__ xl, const int* __restrict__ src,
                               const int* __restrict__ dst, const float* __restrict__ ex,
                               const float* __restrict__ den, float* __restrict__ agg, int E) {
  int wave = threadIdx.x >> 5, lane = threadIdx.x & 31;
  int e = blockIdx.x * 4 + wave;
  if (e >= E) return;
  int s = src[e], d = dst[e];
  for (int h = 0; h < 4; ++h) {
    float alpha = ex[(size_t)e * 4 + h] / den[(size_t)d * 4 + h];
    v4f vj = ((const v4f*)(xl + (size_t)s * 512 + h * 128))[lane];
    float* po = agg + (size_t)d * 512 + h * 128 + lane * 4;
#pragma unroll
    for (int q = 0; q < 4; ++q) atomicAdd(po + q, alpha * vj[q]);
  }
}

// ---------------------------------------------------------------------------
// Utility kernels
// ---------------------------------------------------------------------------
__global__ void k_f32_to_f16(const float* __restrict__ in, h16* __restrict__ out, int n) {
  int i = blockIdx.x * blockDim.x + threadIdx.x;
  if (i < n) out[i] = (h16)in[i];
}
// in: [rows][cols] f32 -> out: [cols][rows] f16  (weight transpose to [K][N])
__global__ void k_f32_to_f16_T(const float* __restrict__ in, h16* __restrict__ out, int rows,
                               int cols) {
  int i = blockIdx.x * blockDim.x + threadIdx.x;
  if (i >= rows * cols) return;
  int r = i / cols, c = i - r * cols;
  out[(size_t)c * rows + r] = (h16)in[i];
}
__global__ void k_zero_f16(h16* __restrict__ p, int n) {
  int i = blockIdx.x * blockDim.x + threadIdx.x;
  if (i < n) p[i] = (h16)0;
}
__global__ void k_zero_f32(float* __restrict__ p, int n) {
  int i = blockIdx.x * blockDim.x + threadIdx.x;
  if (i < n) p[i] = 0.f;
}

// ---------------------------------------------------------------------------
// Host launcher
// ---------------------------------------------------------------------------
extern "C" void kernel_launch(void* const* d_in, const int* in_sizes, int n_in, void* d_out,
                              int out_size, void* d_ws, size_t ws_size, hipStream_t stream) {
  (void)in_sizes; (void)n_in; (void)out_size; (void)ws_size;
  constexpr int NN = 10000, T = 24, IN = 16, HID = 128, HH = 512, E = 120000;
  constexpr int ROWS = NN * T;  // 240000

  const float* x   = (const float*)d_in[0];
  const int*   ei  = (const int*)d_in[1];
  const int*   src = ei;
  const int*   dst = ei + E;

  const float* mlpW[3]  = {(const float*)d_in[2], (const float*)d_in[4], (const float*)d_in[6]};
  const float* mlpB[3]  = {(const float*)d_in[3], (const float*)d_in[5], (const float*)d_in[7]};
  const float* lWih[2]  = {(const float*)d_in[8],  (const float*)d_in[12]};
  const float* lWhh[2]  = {(const float*)d_in[9],  (const float*)d_in[13]};
  const float* lBih[2]  = {(const float*)d_in[10], (const float*)d_in[14]};
  const float* lBhh[2]  = {(const float*)d_in[11], (const float*)d_in[15]};
  const float* gWl[2]   = {(const float*)d_in[16], (const float*)d_in[24]};
  const float* gBl[2]   = {(const float*)d_in[17], (const float*)d_in[25]};
  const float* gWr[2]   = {(const float*)d_in[18], (const float*)d_in[26]};
  const float* gBr[2]   = {(const float*)d_in[19], (const float*)d_in[27]};
  const float* gAtt[2]  = {(const float*)d_in[20], (const float*)d_in[28]};
  const float* gBias[2] = {(const float*)d_in[21], (const float*)d_in[29]};
  const float* gWp[2]   = {(const float*)d_in[22], (const float*)d_in[30]};
  const float* gBp[2]   = {(const float*)d_in[23], (const float*)d_in[31]};
  const float* moW[3] = {(const float*)d_in[32], (const float*)d_in[34], (const float*)d_in[36]};
  const float* moB[3] = {(const float*)d_in[33], (const float*)d_in[35], (const float*)d_in[37]};

  // ---- workspace carve ----
  char*  ws  = (char*)d_ws;
  size_t off = 0;
  auto carve = [&](size_t bytes) -> void* {
    void* p = ws + off;
    off += (bytes + 255) & ~size_t(255);
    return p;
  };
  h16*   xh    = (h16*)carve((size_t)ROWS * IN * 2);
  h16*   bufT0 = (h16*)carve((size_t)ROWS * HID * 2);
  h16*   bufT1 = (h16*)carve((size_t)ROWS * HID * 2);
  h16*   h1h   = (h16*)carve((size_t)NN * HID * 2);
  h16*   h2h   = (h16*)carve((size_t)NN * HID * 2);
  float* c1    = (float*)carve((size_t)NN * HID * 4);
  float* c2    = (float*)carve((size_t)NN * HID * 4);
  h16*   actA  = (h16*)carve((size_t)NN * HID * 2);
  h16*   actB  = (h16*)carve((size_t)NN * HID * 2);
  float* Xl    = (float*)carve((size_t)NN * HH * 4);
  float* Xr    = (float*)carve((size_t)NN * HH * 4);
  float* Agg   = (float*)carve((size_t)NN * HH * 4);
  h16*   Aggh  = (h16*)carve((size_t)NN * HH * 2);
  float* Logit = (float*)carve((size_t)E * 4 * 4);
  float* Ex    = (float*)carve((size_t)E * 4 * 4);
  unsigned* MaxK = (unsigned*)carve((size_t)NN * 4 * 4);
  float*    Den  = (float*)carve((size_t)NN * 4 * 4);
  h16* wMlp[3] = {(h16*)carve(IN * HID * 2), (h16*)carve(HID * HID * 2), (h16*)carve(HID * HID * 2)};
  h16* wIhT[2] = {(h16*)carve(HID * HH * 2), (h16*)carve(HID * HH * 2)};
  h16* wHhT[2] = {(h16*)carve(HID * HH * 2), (h16*)carve(HID * HH * 2)};
  h16* wGl[2]  = {(h16*)carve(HID * HH * 2), (h16*)carve(HID * HH * 2)};
  h16* wGr[2]  = {(h16*)carve(HID * HH * 2), (h16*)carve(HID * HH * 2)};
  h16* wGp[2]  = {(h16*)carve(HH * HID * 2), (h16*)carve(HH * HID * 2)};
  h16* wMo[3] = {(h16*)carve(HID * HID * 2), (h16*)carve(HID * HID * 2), (h16*)carve(HID * 12 * 2)};

  auto cvt = [&](const float* in, h16* out, int n) {
    k_f32_to_f16<<<dim3((n + 255) / 256), dim3(256), 0, stream>>>(in, out, n);
  };
  auto cvtT = [&](const float* in, h16* out, int rows, int cols) {
    int n = rows * cols;
    k_f32_to_f16_T<<<dim3((n + 255) / 256), dim3(256), 0, stream>>>(in, out, rows, cols);
  };
  auto gemm = [&](const h16* A, const h16* B, const float* bias, float* Cf, h16* Ch, int M,
                  int N, int K, int relu) {
    dim3 g((M + TM - 1) / TM, (N + TN - 1) / TN);
    k_gemm_f16<<<g, dim3(128), 0, stream>>>(A, B, bias, Cf, Ch, M, N, K, relu);
  };

  // ---- weight conversion (per-call; deterministic) ----
  cvt(x, xh, ROWS * IN);
  cvt(mlpW[0], wMlp[0], IN * HID);
  cvt(mlpW[1], wMlp[1], HID * HID);
  cvt(mlpW[2], wMlp[2], HID * HID);
  for (int l = 0; l < 2; ++l) {
    cvtT(lWih[l], wIhT[l], HH, HID);  // [512,128] -> [128][512]
    cvtT(lWhh[l], wHhT[l], HH, HID);
    cvt(gWl[l], wGl[l], HID * HH);    // already [K=128][N=512]
    cvt(gWr[l], wGr[l], HID * HH);
    cvt(gWp[l], wGp[l], HH * HID);    // already [K=512][N=128]
  }
  cvt(moW[0], wMo[0], HID * HID);
  cvt(moW[1], wMo[1], HID * HID);
  cvt(moW[2], wMo[2], HID * 12);

  // ---- MLP-in: Linear(16,128), then 2x [Linear(128,128)+ReLU] over 240k rows ----
  gemm(xh, wMlp[0], mlpB[0], nullptr, bufT0, ROWS, HID, IN, 0);
  gemm(bufT0, wMlp[1], mlpB[1], nullptr, bufT1, ROWS, HID, HID, 1);
  gemm(bufT1, wMlp[2], mlpB[2], nullptr, bufT0, ROWS, HID, HID, 1);
  h16* H0 = bufT0;  // [node*24 + t][128] f16

  // ---- LSTM (2 layers, T=24, interleaved) ----
  int nh = NN * HID;
  k_zero_f16<<<dim3((nh + 255) / 256), dim3(256), 0, stream>>>(h1h, nh);
  k_zero_f16<<<dim3((nh + 255) / 256), dim3(256), 0, stream>>>(h2h, nh);
  k_zero_f32<<<dim3((nh + 255) / 256), dim3(256), 0, stream>>>(c1, nh);
  k_zero_f32<<<dim3((nh + 255) / 256), dim3(256), 0, stream>>>(c2, nh);
  dim3 lg((NN + 127) / 128);
  for (int t = 0; t < T; ++t) {
    k_lstm_step<<<lg, dim3(128), 0, stream>>>(H0 + (size_t)t * HID, T * HID, h1h, c1, h1h,
                                              wIhT[0], wHhT[0], lBih[0], lBhh[0], NN);
    k_lstm_step<<<lg, dim3(128), 0, stream>>>(h1h, HID, h2h, c2, h2h, wIhT[1], wHhT[1],
                                              lBih[1], lBhh[1], NN);
  }

  // ---- GAT layers ----
  const h16* act = h2h;
  h16*       gatOut[2] = {actA, actB};
  int nAgg = NN * HH, nSeg = NN * 4;
  for (int l = 0; l < 2; ++l) {
    gemm(act, wGl[l], gBl[l], Xl, nullptr, NN, HH, HID, 0);
    gemm(act, wGr[l], gBr[l], Xr, nullptr, NN, HH, HID, 0);
    k_gat_init<<<dim3((nAgg + 255) / 256), dim3(256), 0, stream>>>(Agg, gBias[l], MaxK, Den,
                                                                   nAgg, nSeg);
    dim3 eg((E + 3) / 4);
    k_edge_logit_max<<<eg, dim3(128), 0, stream>>>(Xl, Xr, src, dst, gAtt[l], Logit, MaxK, E);
    k_edge_exp<<<dim3((E * 4 + 255) / 256), dim3(256), 0, stream>>>(Logit, MaxK, dst, Ex, Den, E);
    k_edge_scatter<<<eg, dim3(128), 0, stream>>>(Xl, src, dst, Ex, Den, Agg, E);
    k_f32_to_f16<<<dim3((nAgg + 255) / 256), dim3(256), 0, stream>>>(Agg, Aggh, nAgg);
    gemm(Aggh, wGp[l], gBp[l], nullptr, gatOut[l], NN, HID, HH, 1);  // ReLU
    act = gatOut[l];
  }

  // ---- MLP-out: 2x [Linear+ReLU] then Linear(128,12) -> d_out (f32) ----
  gemm(actB, wMo[0], moB[0], nullptr, actA, NN, HID, HID, 1);
  gemm(actA, wMo[1], moB[1], nullptr, actB, NN, HID, HID, 1);
  gemm(actB, wMo[2], moB[2], (float*)d_out, nullptr, NN, 12, HID, 0);
}
